// SelfAttention_23605140258966
// MI455X (gfx1250) — matrix operand order repaired and checked by reading, not compiled, not run on hardware
//
#include <hip/hip_runtime.h>

typedef __attribute__((ext_vector_type(16))) _Float16 v16h;
typedef __attribute__((ext_vector_type(8)))  _Float16 v8h;
typedef __attribute__((ext_vector_type(8)))  float    v8f;

constexpr int B_ = 2, T_ = 4096, E_ = 512, H_ = 8, D_ = 64;
constexpr int BH_ = B_ * H_;

#define DEVFN __device__ __forceinline__

DEVFN v8f wmma16x16x32(v16h a, v16h b, v8f c) {
    return __builtin_amdgcn_wmma_f32_16x16x32_f16(
        /*neg_a=*/false, a, /*neg_b=*/false, b,
        /*c_mod=*/(short)0, c, /*reuse_a=*/false, /*reuse_b=*/false);
}

// A fragment, 16x32 (MxK) f16, from row-major src with leading dim ld.
// ISA layout: lane l holds row m=l&15; lanes 0-15: k=[0..7],[16..23];
// lanes 16-31: k=[8..15],[24..31]; pairs packed per VGPR in ascending k.
DEVFN v16h loadA16x32(const _Float16* base, int ld, int lane) {
    const int m    = lane & 15;
    const int koff = (lane >> 4) << 3;          // 0 or 8
    const _Float16* p = base + m * ld + koff;
    v8h lo = *(const v8h*)(p);                  // k = koff .. koff+7
    v8h hi = *(const v8h*)(p + 16);             // k = 16+koff .. 16+koff+7
    return __builtin_shufflevector(lo, hi, 0,1,2,3,4,5,6,7,8,9,10,11,12,13,14,15);
}

// B fragment, 32x16 (KxN) f16 where B[k][n] = src[n][k] (src row-major, ld).
// ISA layout: lane l holds col n=l&15; lanes 0-15: k=[0..7],[16..23];
// lanes 16-31: k=[8..15],[24..31]; packed ascending (matches A).
DEVFN v16h loadB32x16T(const _Float16* base, int ld, int lane) {
    const int n  = lane & 15;
    const int kb = (lane >> 4) << 3;            // 0 or 8
    const _Float16* p = base + n * ld + kb;
    v8h lo = *(const v8h*)(p);
    v8h hi = *(const v8h*)(p + 16);
    return __builtin_shufflevector(lo, hi, 0,1,2,3,4,5,6,7,8,9,10,11,12,13,14,15);
}

// DPP16 butterfly max over the 16-lane half (VALU only, no LDS pipe).
template <int CTRL>
DEVFN float dpp_max_step(float v) {
    int perm = __builtin_amdgcn_update_dpp(0, __builtin_bit_cast(int, v),
                                           CTRL, 0xf, 0xf, true);
    return fmaxf(v, __builtin_bit_cast(float, perm));
}
DEVFN float reduce_max16(float v) {
    v = dpp_max_step<0xB1>(v);   // quad_perm [1,0,3,2] : xor 1
    v = dpp_max_step<0x4E>(v);   // quad_perm [2,3,0,1] : xor 2
    v = dpp_max_step<0x141>(v);  // row_half_mirror     : xor 7
    v = dpp_max_step<0x140>(v);  // row_mirror          : xor 15
    return v;
}

// ---------------- Kernel 0: Wu fp32 -> f16 ----------------
__global__ void __launch_bounds__(256) wu_to_half(const float* __restrict__ Wu,
                                                  _Float16* __restrict__ Wu16) {
    int i = blockIdx.x * 256 + threadIdx.x;
    Wu16[i] = (_Float16)Wu[i];
}

// ---------------- Kernel 1: QKV projections ----------------
// Q,K: [bh][t][64] f16 (Q pre-scaled by 1/sqrt(512)); Vt: [bh][64][t] f16.
__global__ void __launch_bounds__(256) qkv_kernel(const float* __restrict__ x,
                                                  const float* __restrict__ Wq,
                                                  const float* __restrict__ Wk,
                                                  const float* __restrict__ Wv,
                                                  _Float16* __restrict__ Q,
                                                  _Float16* __restrict__ K,
                                                  _Float16* __restrict__ Vt) {
    __shared__ float smem[4 * 64 * 64];   // xs | wq | wk | wv  (64 KB)
    float* xs = smem;
    float* wq = smem + 4096;
    float* wk = smem + 8192;
    float* wv = smem + 12288;

    const int tid   = threadIdx.x;
    const int tiles = T_ / 64;
    const int tileT = blockIdx.x % tiles;
    const int bh    = blockIdx.x / tiles;
    const int b     = bh / H_, h = bh % H_;
    const int t0    = tileT * 64;

    for (int i = tid; i < 4096; i += 256) {
        wq[i] = Wq[i]; wk[i] = Wk[i]; wv[i] = Wv[i];
        int tok = i >> 6, d = i & 63;
        xs[i] = x[(size_t)(b * T_ + t0 + tok) * E_ + h * D_ + d];
    }
    __syncthreads();

    const float qscale = 0.044194173824159216f;   // 1/sqrt(512)
    for (int p = tid; p < 4096; p += 256) {
        int tok = p >> 6, e = p & 63;
        float sq = 0.f, sk = 0.f, sv = 0.f;
        #pragma unroll 8
        for (int d = 0; d < 64; ++d) {
            float xv = xs[tok * 64 + d];
            sq += xv * wq[e * 64 + d];
            sk += xv * wk[e * 64 + d];
            sv += xv * wv[e * 64 + d];
        }
        size_t row = (size_t)bh * T_ + (t0 + tok);
        Q[row * D_ + e] = (_Float16)(sq * qscale);
        K[row * D_ + e] = (_Float16)sk;
        Vt[((size_t)bh * D_ + e) * T_ + (t0 + tok)] = (_Float16)sv;
    }
}

// ---------------- Kernel 2: flash attention ----------------
// 4 waves/block; each wave owns 16 query rows, 64-key KV tiles.
// Row-sum of P is computed on the XDL pipe (ones-matrix WMMA);
// row-max uses a DPP16 butterfly (no LDS pipe traffic).
__global__ void __launch_bounds__(128) attn_kernel(const _Float16* __restrict__ Q,
                                                   const _Float16* __restrict__ K,
                                                   const _Float16* __restrict__ Vt,
                                                   _Float16* __restrict__ attn) {
    __shared__ __align__(32) _Float16 pbuf[4][16 * 64];   // per-wave P tile

    const int lane  = threadIdx.x & 31;
    const int wave  = threadIdx.x >> 5;
    const int tiles = T_ / 64;
    const int bh    = blockIdx.x / tiles;
    const int tileQ = blockIdx.x % tiles;
    const int q0    = tileQ * 64 + wave * 16;
    const int b     = bh / H_, h = bh % H_;

    const _Float16* Qb = Q  + (size_t)bh * T_ * D_;
    const _Float16* Kb = K  + (size_t)bh * T_ * D_;
    const _Float16* Vb = Vt + (size_t)bh * D_ * T_;

    // Q fragments for this wave's 16 rows, k=0..31 and 32..63 (held all loop).
    v16h aq0 = loadA16x32(Qb + (size_t)q0 * D_,      D_, lane);
    v16h aq1 = loadA16x32(Qb + (size_t)q0 * D_ + 32, D_, lane);

    v16h ones;
    #pragma unroll
    for (int i = 0; i < 16; ++i) ones[i] = (_Float16)1.0f;

    v8f o0 = {}, o1 = {}, o2 = {}, o3 = {}, ls = {};
    float mrow[8];
    #pragma unroll
    for (int r = 0; r < 8; ++r) mrow[r] = -3.0e38f;

    _Float16* pb = pbuf[wave];
    const int mMine = (lane >> 4) << 3;     // 0 or 8: row base for my C rows
    const int nMine = lane & 15;

    for (int kv = 0; kv < T_; kv += 64) {
        // ---- S = Q * K^T : 16 queries x 64 keys = four 16x16 C tiles ----
        v8f s[4];
        #pragma unroll
        for (int i = 0; i < 4; ++i) {
            v16h b0 = loadB32x16T(Kb + (size_t)(kv + 16 * i) * D_,      D_, lane);
            v16h b1 = loadB32x16T(Kb + (size_t)(kv + 16 * i) * D_ + 32, D_, lane);
            v8f t = {};
            t = wmma16x16x32(aq0, b0, t);
            t = wmma16x16x32(aq1, b1, t);
            s[i] = t;
        }

        // prefetch next KV tile while softmax runs (global_prefetch_b8)
        {
            const int kvn = (kv + 64) & (T_ - 1);
            __builtin_prefetch(Kb + (size_t)kvn * D_ + lane * 128, 0, 1);
            __builtin_prefetch(Vb + (size_t)lane * T_ + kvn, 0, 1);
            __builtin_prefetch(Vb + (size_t)(lane + 32) * T_ + kvn, 0, 1);
        }

        // ---- online softmax: DPP row-max, exp, spill P to LDS ----
        float scale[8];
        #pragma unroll
        for (int r = 0; r < 8; ++r) {
            float mx = fmaxf(fmaxf(s[0][r], s[1][r]), fmaxf(s[2][r], s[3][r]));
            mx = reduce_max16(mx);
            float mnew = fmaxf(mrow[r], mx);
            scale[r]   = __expf(mrow[r] - mnew);
            mrow[r]    = mnew;
            _Float16* prow = pb + (r + mMine) * 64 + nMine;
            prow[0]  = (_Float16)__expf(s[0][r] - mnew);
            prow[16] = (_Float16)__expf(s[1][r] - mnew);
            prow[32] = (_Float16)__expf(s[2][r] - mnew);
            prow[48] = (_Float16)__expf(s[3][r] - mnew);
        }
        #pragma unroll
        for (int r = 0; r < 8; ++r) {
            o0[r] *= scale[r]; o1[r] *= scale[r];
            o2[r] *= scale[r]; o3[r] *= scale[r];
            ls[r] *= scale[r];
        }

        // ---- A fragments of P (k = key within tile) ----
        v16h ap0 = loadA16x32(pb,      64, lane);
        v16h ap1 = loadA16x32(pb + 32, 64, lane);

        // row-sum of P on the matrix pipe: ls += P @ ones
        ls = wmma16x16x32(ap0, ones, ls);
        ls = wmma16x16x32(ap1, ones, ls);

        // ---- O += P * V ----
        #pragma unroll
        for (int i = 0; i < 4; ++i) {
            v16h bv0 = loadB32x16T(Vb + (size_t)(16 * i) * T_ + kv,      T_, lane);
            v16h bv1 = loadB32x16T(Vb + (size_t)(16 * i) * T_ + kv + 32, T_, lane);
            v8f* o = (i == 0) ? &o0 : (i == 1) ? &o1 : (i == 2) ? &o2 : &o3;
            *o = wmma16x16x32(ap0, bv0, *o);
            *o = wmma16x16x32(ap1, bv1, *o);
        }
    }

    // ---- finalize: O /= rowsum (ls holds it per-lane already) ----
    #pragma unroll
    for (int r = 0; r < 8; ++r) {
        float inv = 1.0f / ls[r];
        int m = r + mMine;
        size_t base = (size_t)(b * T_ + q0 + m) * E_ + h * D_;
        attn[base +  0 + nMine] = (_Float16)(o0[r] * inv);
        attn[base + 16 + nMine] = (_Float16)(o1[r] * inv);
        attn[base + 32 + nMine] = (_Float16)(o2[r] * inv);
        attn[base + 48 + nMine] = (_Float16)(o3[r] * inv);
    }
}

// ---------------- Kernel 3: out = attn @ Wu^T + bu ----------------
__global__ void __launch_bounds__(256) proj_kernel(const _Float16* __restrict__ attn,
                                                   const _Float16* __restrict__ Wu16,
                                                   const float* __restrict__ bu,
                                                   float* __restrict__ out) {
    const int lane = threadIdx.x & 31;
    const int wave = threadIdx.x >> 5;
    const int gtile   = blockIdx.x * 8 + wave;   // 4096 wave-tiles total
    const int rowTile = gtile >> 3;              // 512 row tiles of 16
    const int colG    = gtile & 7;               // 8 column groups of 64
    const int row0 = rowTile * 16;
    const int col0 = colG * 64;

    v8f c0 = {}, c1 = {}, c2 = {}, c3 = {};
    for (int k = 0; k < E_; k += 32) {
        v16h a = loadA16x32(attn + (size_t)row0 * E_ + k, E_, lane);
        c0 = wmma16x16x32(a, loadB32x16T(Wu16 + (size_t)(col0 +  0) * E_ + k, E_, lane), c0);
        c1 = wmma16x16x32(a, loadB32x16T(Wu16 + (size_t)(col0 + 16) * E_ + k, E_, lane), c1);
        c2 = wmma16x16x32(a, loadB32x16T(Wu16 + (size_t)(col0 + 32) * E_ + k, E_, lane), c2);
        c3 = wmma16x16x32(a, loadB32x16T(Wu16 + (size_t)(col0 + 48) * E_ + k, E_, lane), c3);
    }

    const int nMine = lane & 15;
    const int mBase = (lane >> 4) << 3;
    #pragma unroll
    for (int r = 0; r < 8; ++r) {
        int m = r + mBase;
        size_t o = (size_t)(row0 + m) * E_ + col0;
        out[o +  0 + nMine] = c0[r] + bu[col0 +  0 + nMine];
        out[o + 16 + nMine] = c1[r] + bu[col0 + 16 + nMine];
        out[o + 32 + nMine] = c2[r] + bu[col0 + 32 + nMine];
        out[o + 48 + nMine] = c3[r] + bu[col0 + 48 + nMine];
    }
}

extern "C" void kernel_launch(void* const* d_in, const int* in_sizes, int n_in,
                              void* d_out, int out_size, void* d_ws, size_t ws_size,
                              hipStream_t stream) {
    (void)in_sizes; (void)n_in; (void)out_size; (void)ws_size;
    const float* x  = (const float*)d_in[0];
    const float* Wq = (const float*)d_in[1];
    const float* Wk = (const float*)d_in[2];
    const float* Wv = (const float*)d_in[3];
    const float* Wu = (const float*)d_in[4];
    const float* bu = (const float*)d_in[5];
    float* out = (float*)d_out;

    char* ws = (char*)d_ws;
    const size_t szQKV = (size_t)BH_ * T_ * D_ * sizeof(_Float16);   // 8 MB each
    _Float16* Q    = (_Float16*)(ws);
    _Float16* K    = (_Float16*)(ws + szQKV);
    _Float16* Vt   = (_Float16*)(ws + 2 * szQKV);
    _Float16* attn = (_Float16*)(ws + 3 * szQKV);                    // [b*t][E]
    _Float16* Wu16 = (_Float16*)(ws + 4 * szQKV);

    wu_to_half<<<(E_ * E_) / 256, 256, 0, stream>>>(Wu, Wu16);
    qkv_kernel<<<BH_ * (T_ / 64), 256, 0, stream>>>(x, Wq, Wk, Wv, Q, K, Vt);
    attn_kernel<<<BH_ * (T_ / 64), 128, 0, stream>>>(Q, K, Vt, attn);
    proj_kernel<<<(B_ * T_ / 16) * 8 / 8, 256, 0, stream>>>(attn, Wu16, bu, out);
}
